// SpecAPNBA_4758823764001
// MI455X (gfx1250) — matrix operand order repaired and checked by reading, not compiled, not run on hardware
//
#include <hip/hip_runtime.h>

typedef __attribute__((ext_vector_type(16))) _Float16 v16h;
typedef __attribute__((ext_vector_type(8)))  float    v8f;

#define C_DIM 200
#define HW    225
#define FCH   12
#define FR    6
#define KPAD  224   // 7 * 32 (K padded; rows 200..223 zeroed in LDS)
#define NT    15    // ceil(225/16) N-tiles
#define NPAD  240
#define LN_EPS 1e-5f

// LDS carve-up (floats)
#define XS_N    50688            // 224*225 = 50400 (+ slack for N-overhang reads)
#define AF_N    5376             // 3*7*32*16 f16 = 10752 halves = 5376 float slots
#define VBUF_N  (FCH*NPAD)       // 2880
#define KQS_N   NPAD             // 240
#define BIAS_N  40               // 3*12 biases (k,q,v)
#define VMACC_N 16
#define VS_N    16
#define T1_N    8
#define T2_N    C_DIM
#define T3_N    C_DIM
#define LNG_N   C_DIM
#define LNB_N   C_DIM
#define STATS_N 8
#define MU_N    232
#define RSTD_N  232
#define SMEM_FLOATS (XS_N+AF_N+VBUF_N+KQS_N+BIAS_N+VMACC_N+VS_N+T1_N+T2_N+T3_N+LNG_N+LNB_N+STATS_N+MU_N+RSTD_N)

__global__ __launch_bounds__(256) void apnba_fused_kernel(
    const float* __restrict__ X,
    const float* __restrict__ w11, const float* __restrict__ b11,
    const float* __restrict__ w21, const float* __restrict__ b21,
    const float* __restrict__ w31, const float* __restrict__ b31,
    const float* __restrict__ w41, const float* __restrict__ b41,
    const float* __restrict__ w42, const float* __restrict__ b42,
    const float* __restrict__ ln_g, const float* __restrict__ ln_b,
    float* __restrict__ out)
{
    extern __shared__ float smem[];
    float*    Xs    = smem;                       // [224*225] row-major [k][n], rows 200+ zero
    _Float16* Af    = (_Float16*)(Xs + XS_N);     // [3][7][32][16] f16 A-fragments
    float*    vbuf  = Xs + XS_N + AF_N;           // [12][NPAD]
    float*    kqs   = vbuf  + VBUF_N;             // [NPAD]
    float*    biasS = kqs   + KQS_N;              // [36]: b11,b21,b31
    float*    vmacc = biasS + BIAS_N;             // [12]
    float*    vs    = vmacc + VMACC_N;            // [12]
    float*    t1    = vs    + VS_N;               // [6]
    float*    t2    = t1    + T1_N;               // [200]
    float*    t3    = t2    + T2_N;               // [200]
    float*    lng   = t3    + T3_N;               // [200]
    float*    lnb   = lng   + LNG_N;              // [200]
    float*    stats = lnb   + LNB_N;              // [5]
    float*    mu    = stats + STATS_N;            // [225]
    float*    rstd  = mu    + MU_N;               // [225]

    const int tid = threadIdx.x;
    const int b   = blockIdx.x;

    // ---- Phase 0a: async-stage X[b] into LDS (CDNA5 global->LDS DMA path) ----
    {
        const unsigned lds_base = (unsigned)(uintptr_t)(void*)Xs;  // aperture: addr[31:0] == LDS offset
        const float* gbase = X + (size_t)b * (C_DIM * HW);
        for (int i = tid; i < (C_DIM * HW) / 4; i += 256) {        // 11250 x 16B
            unsigned      laddr = lds_base + 16u * (unsigned)i;
            const float4* gaddr = (const float4*)gbase + i;
            asm volatile("global_load_async_to_lds_b128 %0, %1, off"
                         :: "v"(laddr), "v"(gaddr) : "memory");
        }
    }

    // ---- Phase 0b: zero K-pad rows + slack; swizzle weights to f16 A-frags ----
    if (tid < 12) vmacc[tid] = 0.f;
    if (tid < 8)  stats[tid] = 0.f;
    for (int i = C_DIM * HW + tid; i < XS_N; i += 256) Xs[i] = 0.f;

    // A-fragment order (16x32 f16): row M = lane&15; lanes<16 K={0..7,16..23}; lanes>=16 K={8..15,24..31}
    for (int idx = tid; idx < 3 * 7 * 32 * 16; idx += 256) {
        int g    = idx / (7 * 32 * 16);
        int r    = idx % (7 * 32 * 16);
        int kt   = r / (32 * 16);
        int r2   = r % (32 * 16);
        int ln   = r2 / 16;
        int i    = r2 % 16;
        int m    = ln & 15;
        int koff = (ln >> 4) * 8;
        int k    = kt * 32 + ((i < 8) ? (koff + i) : (16 + koff + (i - 8)));
        const float* Wg = (g == 0) ? w11 : (g == 1) ? w21 : w31;
        float val = (m < FCH && k < C_DIM) ? Wg[m * C_DIM + k] : 0.f;
        Af[idx] = (_Float16)val;
    }
    if (tid < 36) {
        int g = tid / 12, f = tid % 12;
        const float* Bg = (g == 0) ? b11 : (g == 1) ? b21 : b31;
        biasS[tid] = Bg[f];
    }
    for (int i = tid; i < C_DIM; i += 256) { lng[i] = ln_g[i]; lnb[i] = ln_b[i]; }

    asm volatile("s_wait_asynccnt 0x0" ::: "memory");
    __syncthreads();

    // ---- Phase 1: WMMA GEMMs  k/q/v[16, 225] = W[16,224] @ Xs[224,225] ----
    const int lane    = tid & 31;
    const int wave    = tid >> 5;      // 0..7
    const int halfsel = lane >> 4;     // 0 | 1
    const int lm      = lane & 15;

    for (int t = wave; t < NT; t += 8) {
        const int n0 = t * 16;
        const int n  = n0 + lm;        // may exceed 224; reads stay in-bounds, results discarded
        v8f ack = {}; v8f acq = {}; v8f acv = {};
        const float* Xcol = Xs + n;

        for (int kt = 0; kt < 7; ++kt) {
            const int fb = (kt * 32 + lane) * 16;        // lane's 16-half chunk within a kt slab
            v16h ak = *(const v16h*)(Af + 0 * (7 * 32 * 16) + fb);
            v16h aq = *(const v16h*)(Af + 1 * (7 * 32 * 16) + fb);
            v16h av = *(const v16h*)(Af + 2 * (7 * 32 * 16) + fb);

            // B fragment (32x16 f16): col N=lm; lanes<16 K=kb+{0..7,16..23}, lanes>=16 K=kb+{8..15,24..31}
            const int kcb = kt * 32 + halfsel * 8;
            v16h bx;
            #pragma unroll
            for (int i = 0; i < 16; ++i)
                bx[i] = (_Float16)Xcol[(kcb + ((i < 8) ? i : (i + 8))) * HW];

            ack = __builtin_amdgcn_wmma_f32_16x16x32_f16(false, ak, false, bx, (short)0, ack, false, false);
            acq = __builtin_amdgcn_wmma_f32_16x16x32_f16(false, aq, false, bx, (short)0, acq, false, false);
            acv = __builtin_amdgcn_wmma_f32_16x16x32_f16(false, av, false, bx, (short)0, acv, false, false);
        }
        // C/D layout: vgpr r, lane L -> row M = r + 8*(L/16), col = n0 + L%16
        float part = 0.f;
        #pragma unroll
        for (int r = 0; r < 8; ++r) {
            int M = r + 8 * halfsel;
            if (M < FCH) {
                float kv = ack[r] + biasS[M];
                float qv = acq[r] + biasS[12 + M];
                part += kv * qv;
                vbuf[M * NPAD + n0 + lm] = acv[r] + biasS[24 + M];
            }
        }
        part += __shfl_xor(part, 16);          // combine rows 0-7 with 8-15
        if (halfsel == 0 && n < HW) kqs[n] = part;
    }
    __syncthreads();

    // ---- Phase 2: scrambled chunk means vm[i] = mean_{t in [i*225,(i+1)*225)} v[t%12][t/12] ----
    for (int t = tid; t < FCH * HW; t += 256) {
        int chunk = t / HW;
        int f     = t % FCH;
        int sp    = t / FCH;
        atomicAdd(&vmacc[chunk], vbuf[f * NPAD + sp]);
    }
    __syncthreads();

    // ---- Phase 3: softmax -> vs ----
    if (tid == 0) {
        float vmv[FCH]; float m = -1e30f;
        for (int i = 0; i < FCH; ++i) { vmv[i] = vmacc[i] * (1.f / (float)HW); m = fmaxf(m, vmv[i]); }
        float s = 0.f;
        for (int i = 0; i < FCH; ++i) { float e = __expf(vmv[i] - m); vs[i] = e; s += e; }
        float inv = 1.f / s;
        for (int i = 0; i < FCH; ++i) vs[i] *= inv;
    }
    __syncthreads();

    // ---- Phase 4: t1 = w41@vs ; t2 = w42@t1 ; t3 = w42@b41 + b42 ; moments ----
    if (tid < FR) {
        float acc = 0.f;
        for (int l = 0; l < FCH; ++l) acc += w41[tid * FCH + l] * vs[l];
        t1[tid] = acc;
    }
    __syncthreads();
    if (tid < C_DIM) {
        float a2 = 0.f, a3 = 0.f;
        for (int r = 0; r < FR; ++r) {
            float wv = w42[tid * FR + r];
            a2 += wv * t1[r];
            a3 += wv * b41[r];
        }
        a3 += b42[tid];
        t2[tid] = a2; t3[tid] = a3;
        atomicAdd(&stats[0], a2);
        atomicAdd(&stats[1], a3);
        atomicAdd(&stats[2], a2 * a2);
        atomicAdd(&stats[3], a3 * a3);
        atomicAdd(&stats[4], a2 * a3);
    }
    __syncthreads();

    // ---- Phase 5: closed-form LayerNorm stats per spatial position ----
    if (tid < HW) {
        const float invC = 1.f / (float)C_DIM;
        float kq  = kqs[tid];
        float m2  = stats[0] * invC, m3  = stats[1] * invC;
        float e22 = stats[2] * invC, e33 = stats[3] * invC, e23 = stats[4] * invC;
        float muv = kq * m2 + m3;
        float ex2 = kq * kq * e22 + 2.f * kq * e23 + e33;
        mu[tid]   = muv;
        rstd[tid] = rsqrtf(ex2 - muv * muv + LN_EPS);
    }
    __syncthreads();

    // ---- Phase 6: gate = sigmoid(relu(LN(x2))) ; out = gate * X (X from LDS) ----
    float* outb = out + (size_t)b * (C_DIM * HW);
    for (int i = tid; i < C_DIM * HW; i += 256) {
        int c = i / HW;
        int n = i - c * HW;
        float x2   = kqs[n] * t2[c] + t3[c];
        float xn   = (x2 - mu[n]) * rstd[n] * lng[c] + lnb[c];
        float rl   = fmaxf(xn, 0.f);
        float gate = 1.f / (1.f + __expf(-rl));
        outb[i] = gate * Xs[i];
    }
}

extern "C" void kernel_launch(void* const* d_in, const int* in_sizes, int n_in,
                              void* d_out, int out_size, void* d_ws, size_t ws_size,
                              hipStream_t stream) {
    (void)in_sizes; (void)n_in; (void)d_ws; (void)ws_size; (void)out_size;
    const float* X    = (const float*)d_in[0];
    const float* w11  = (const float*)d_in[1];
    const float* b11  = (const float*)d_in[2];
    const float* w21  = (const float*)d_in[3];
    const float* b21  = (const float*)d_in[4];
    const float* w31  = (const float*)d_in[5];
    const float* b31  = (const float*)d_in[6];
    const float* w41  = (const float*)d_in[7];
    const float* b41  = (const float*)d_in[8];
    const float* w42  = (const float*)d_in[9];
    const float* b42  = (const float*)d_in[10];
    const float* ln_g = (const float*)d_in[11];
    const float* ln_b = (const float*)d_in[12];
    float* out = (float*)d_out;

    const size_t smem_bytes = (size_t)SMEM_FLOATS * sizeof(float);
    hipFuncSetAttribute((const void*)apnba_fused_kernel,
                        hipFuncAttributeMaxDynamicSharedMemorySize, (int)smem_bytes);

    apnba_fused_kernel<<<dim3(1024), dim3(256), smem_bytes, stream>>>(
        X, w11, b11, w21, b21, w31, b31, w41, b41, w42, b42, ln_g, ln_b, out);
}
